// MultiHeadSelfAttention_75316546503234
// MI455X (gfx1250) — matrix operand order repaired and checked by reading, not compiled, not run on hardware
//
#include <hip/hip_runtime.h>
#include <math.h>

#define EMBED 1024
#define NHEAD 16
#define HDIM  64
#define SEQ   2048
#define BATCH 4
#define ROWS  (BATCH * SEQ)   // 8192

typedef __attribute__((ext_vector_type(16))) _Float16     v16h;
typedef __attribute__((ext_vector_type(8)))  float        v8f;
typedef __attribute__((ext_vector_type(4)))  unsigned int v4u;
typedef __attribute__((ext_vector_type(4)))  int          i4;

#if __has_builtin(__builtin_amdgcn_global_load_async_to_lds_b128)
#define HAVE_ASYNC 1
#endif

// 16-byte global -> LDS copy (per lane). Async path uses the CDNA5
// GLOBAL_LOAD_ASYNC_TO_LDS_B128 instruction (tracked by ASYNCcnt).
__device__ __forceinline__ void async_cp16(const void* g, void* l) {
#ifdef HAVE_ASYNC
    __builtin_amdgcn_global_load_async_to_lds_b128(
        (__attribute__((address_space(1))) i4*)g,
        (__attribute__((address_space(3))) i4*)l, 0, 0);
#else
    *(v4u*)l = *(const v4u*)g;
#endif
}

template <int N>
__device__ __forceinline__ void async_wait() {
#ifdef HAVE_ASYNC
#if __has_builtin(__builtin_amdgcn_s_wait_asynccnt)
    __builtin_amdgcn_s_wait_asynccnt(N);
#else
    asm volatile("s_wait_asynccnt %0" ::"i"(N) : "memory");
#endif
#endif
}

__device__ __forceinline__ v8f zero8() {
    v8f z;
#pragma unroll
    for (int i = 0; i < 8; ++i) z[i] = 0.0f;
    return z;
}

// A-fragment (16x32 f16): base at tile origin (row0,k0), row stride ld.
// lane<16: row=lane, halves={K0..7,K16..23}; lane>=16: row=lane-16, halves={K8..15,K24..31}.
__device__ __forceinline__ v16h load_a(const _Float16* base, int ld) {
    const int lane = threadIdx.x & 31;
    const _Float16* p = base + (lane & 15) * ld + (lane >> 4) * 8;
    v16h a;
#pragma unroll
    for (int i = 0; i < 8; ++i) { a[i] = p[i]; a[i + 8] = p[i + 16]; }
    return a;
}

// B-fragment (32x16 f16) from [N][K]-major buffer; base at (n0,k0), ld = leading dim.
// lane<16: col=lane, halves={K0..7,K16..23}; lane>=16: col=lane-16, halves={K8..15,K24..31}.
__device__ __forceinline__ v16h load_b(const _Float16* base, int ld) {
    const int lane = threadIdx.x & 31;
    const _Float16* p = base + (lane & 15) * ld + (lane >> 4) * 8;
    v16h b;
#pragma unroll
    for (int i = 0; i < 8; ++i) { b[i] = p[i]; b[i + 8] = p[i + 16]; }
    return b;
}

__device__ __forceinline__ v8f wmma_f16(v16h a, v16h b, v8f c) {
    return __builtin_amdgcn_wmma_f32_16x16x32_f16(false, a, false, b, (short)0, c, false, false);
}

// ---------------- prep: x + pe -> f16 ----------------
__global__ __launch_bounds__(256) void k_prep_x(const float* __restrict__ x,
                                                const float* __restrict__ pe,
                                                _Float16* __restrict__ xp16) {
    int i = blockIdx.x * 256 + threadIdx.x;
    int col = i & (EMBED - 1);
    int row = i >> 10;
    int s = row & (SEQ - 1);
    xp16[i] = (_Float16)(x[i] + pe[s * EMBED + col]);
}

// ---------------- prep: weights [K][N] f32 -> [N][K] f16 ----------------
__global__ __launch_bounds__(256) void k_prep_w(const float* __restrict__ w0, const float* __restrict__ w1,
                                                const float* __restrict__ w2, const float* __restrict__ w3,
                                                _Float16* __restrict__ t0, _Float16* __restrict__ t1,
                                                _Float16* __restrict__ t2, _Float16* __restrict__ t3) {
    const float* w; _Float16* t;
    switch (blockIdx.z) {
        case 0:  w = w0; t = t0; break;
        case 1:  w = w1; t = t1; break;
        case 2:  w = w2; t = t2; break;
        default: w = w3; t = t3; break;
    }
    int i = blockIdx.x * 256 + threadIdx.x;
    int n = i & (EMBED - 1);
    int k = i >> 10;
    t[n * EMBED + k] = (_Float16)w[k * EMBED + n];
}

// ---------------- fused QKV projection GEMM (LDS double-buffered) ----------------
// grid (ROWS/128, EMBED/64, 3); 8 waves; block tile 128(M) x 64(N); k-step 32.
__global__ __launch_bounds__(256) void k_qkv(const _Float16* __restrict__ xp16,
                                             const _Float16* __restrict__ wTq,
                                             const _Float16* __restrict__ wTk,
                                             const _Float16* __restrict__ wTv,
                                             const float* __restrict__ bq,
                                             const float* __restrict__ bk,
                                             const float* __restrict__ bv,
                                             _Float16* __restrict__ qg,
                                             _Float16* __restrict__ kg,
                                             _Float16* __restrict__ vg) {
    __shared__ _Float16 sA[2][128][40];
    __shared__ _Float16 sB[2][64][40];

    const int wave = threadIdx.x >> 5;
    const int lane = threadIdx.x & 31;
    const int mb = blockIdx.x * 128;
    const int n0 = blockIdx.y * 64;
    const int z  = blockIdx.z;

    const _Float16* wT  = (z == 0) ? wTq : (z == 1) ? wTk : wTv;
    const float*    bia = (z == 0) ? bq  : (z == 1) ? bk  : bv;
    _Float16*       dst = (z == 0) ? qg  : (z == 1) ? kg  : vg;
    const float outscale = (z == 0) ? 0.125f : 1.0f;   // fold 1/sqrt(Hd) into Q

    auto stage = [&](int buf, int kk) {
        const int t = threadIdx.x;
#pragma unroll
        for (int i = 0; i < 2; ++i) {          // A: 128x32 halves = 512 16B chunks
            int chunk = t + i * 256;
            int row = chunk >> 2;
            int off = (chunk & 3) * 8;
            async_cp16(xp16 + (size_t)(mb + row) * EMBED + kk + off, &sA[buf][row][off]);
        }
        {                                      // B: 64x32 halves = 256 16B chunks
            int row = t >> 2;
            int off = (t & 3) * 8;
            async_cp16(wT + (size_t)(n0 + row) * EMBED + kk + off, &sB[buf][row][off]);
        }
    };

    v8f c[4];
#pragma unroll
    for (int j = 0; j < 4; ++j) c[j] = zero8();

    // all fragments live before the WMMA burst -> single DS wait per k-step
    auto compute = [&](int buf) {
        v16h a = load_a(&sA[buf][wave * 16][0], 40);
        v16h bf[4];
#pragma unroll
        for (int j = 0; j < 4; ++j) bf[j] = load_b(&sB[buf][j * 16][0], 40);
#pragma unroll
        for (int j = 0; j < 4; ++j) c[j] = wmma_f16(a, bf[j], c[j]);
    };

    stage(0, 0);
    for (int kk = 0; kk < EMBED; kk += 64) {
        stage(1, kk + 32);
        async_wait<3>();
        __syncthreads();
        compute(0);
        __syncthreads();
        if (kk + 64 < EMBED) { stage(0, kk + 64); async_wait<3>(); }
        else                 { async_wait<0>(); }
        __syncthreads();
        compute(1);
        __syncthreads();
    }

    const int cb = lane & 15;
    const int rh = (lane >> 4) * 8;
    const int m0 = mb + wave * 16;
#pragma unroll
    for (int j = 0; j < 4; ++j) {
        int n = n0 + j * 16 + cb;
        int h = n >> 6, d = n & 63;
        float bvv = bia[n];
#pragma unroll
        for (int r = 0; r < 8; ++r) {
            int m = m0 + r + rh;
            int bb = m >> 11, s = m & (SEQ - 1);
            float val = (c[j][r] + bvv) * outscale;
            dst[(((size_t)(bb * NHEAD + h) * SEQ) + s) * HDIM + d] = (_Float16)val;
        }
    }
}

// ---------------- flash attention (double-buffered K/V tiles) ----------------
// grid (SEQ/128, BATCH*NHEAD); each wave: 16 queries; 32-key steps.
__global__ __launch_bounds__(256) void k_attn(const _Float16* __restrict__ qg,
                                              const _Float16* __restrict__ kg,
                                              const _Float16* __restrict__ vg,
                                              _Float16* __restrict__ og) {
    __shared__ _Float16 sK[2][32][72];     // [key][d], padded
    __shared__ _Float16 sVt[2][64][40];    // [d][key], padded
    __shared__ _Float16 sP[8][16][32];     // per-wave P relayout tile

    const int wave = threadIdx.x >> 5;
    const int lane = threadIdx.x & 31;
    const int bh = blockIdx.y;
    const int b = bh >> 4, h = bh & 15;
    const int q0 = blockIdx.x * 128 + wave * 16;

    const _Float16* qb = qg + ((size_t)bh * SEQ + q0) * HDIM;
    const _Float16* kb = kg + (size_t)bh * SEQ * HDIM;
    const _Float16* vb = vg + (size_t)bh * SEQ * HDIM;

    const int tkey = threadIdx.x >> 3;        // 0..31
    const int toff = (threadIdx.x & 7) * 8;   // 0..56

    auto stageK = [&](int buf, int kt) {
        async_cp16(kb + (size_t)(kt + tkey) * HDIM + toff, &sK[buf][tkey][toff]);
    };
    auto loadVreg = [&](int kt) -> v4u {
        return *(const v4u*)(vb + (size_t)(kt + tkey) * HDIM + toff);
    };
    auto storeV = [&](int buf, v4u vr) {
        union { v4u u; _Float16 hh[8]; } vv; vv.u = vr;
#pragma unroll
        for (int i = 0; i < 8; ++i) sVt[buf][toff + i][tkey] = vv.hh[i];
    };

    v16h aq0 = load_a(qb + 0, HDIM);
    v16h aq1 = load_a(qb + 32, HDIM);

    v8f acc[4];
#pragma unroll
    for (int j = 0; j < 4; ++j) acc[j] = zero8();
    float mrow[8], lrow[8];
#pragma unroll
    for (int r = 0; r < 8; ++r) { mrow[r] = -1e30f; lrow[r] = 0.0f; }

    // prologue: fill buffer 0
    {
        v4u v0 = loadVreg(0);
        stageK(0, 0);
        storeV(0, v0);
        async_wait<0>();
        __syncthreads();
    }

    for (int kt = 0; kt < SEQ; kt += 32) {
        const int buf = (kt >> 5) & 1;
        const bool more = (kt + 32) < SEQ;

        v4u vn;
        if (more) { vn = loadVreg(kt + 32); stageK(buf ^ 1, kt + 32); }

        // ---- scores: 16 q x 32 keys (load all 4 B-frags, then WMMA burst) ----
        v8f s0 = zero8(), s1 = zero8();
        {
            v16h bk0 = load_b(&sK[buf][0][0], 72);
            v16h bk1 = load_b(&sK[buf][0][32], 72);
            v16h bk2 = load_b(&sK[buf][16][0], 72);
            v16h bk3 = load_b(&sK[buf][16][32], 72);
            s0 = wmma_f16(aq0, bk0, s0);
            s0 = wmma_f16(aq1, bk1, s0);
            s1 = wmma_f16(aq0, bk2, s1);
            s1 = wmma_f16(aq1, bk3, s1);
        }

        // ---- online softmax (row stats per lane half) ----
        float rmax[8];
#pragma unroll
        for (int r = 0; r < 8; ++r) rmax[r] = fmaxf(s0[r], s1[r]);
#pragma unroll
        for (int off = 1; off < 16; off <<= 1)
#pragma unroll
            for (int r = 0; r < 8; ++r)
                rmax[r] = fmaxf(rmax[r], __shfl_xor(rmax[r], off, 32));

        float rsum[8];
#pragma unroll
        for (int r = 0; r < 8; ++r) {
            float mnew = fmaxf(mrow[r], rmax[r]);
            float sc = __expf(mrow[r] - mnew);
            mrow[r] = mnew;
            float p0 = __expf(s0[r] - mnew);
            float p1 = __expf(s1[r] - mnew);
            s0[r] = p0; s1[r] = p1;
            rsum[r] = p0 + p1;
            lrow[r] *= sc;
            acc[0][r] *= sc; acc[1][r] *= sc; acc[2][r] *= sc; acc[3][r] *= sc;
        }
#pragma unroll
        for (int off = 1; off < 16; off <<= 1)
#pragma unroll
            for (int r = 0; r < 8; ++r)
                rsum[r] += __shfl_xor(rsum[r], off, 32);
#pragma unroll
        for (int r = 0; r < 8; ++r) lrow[r] += rsum[r];

        // ---- P f32 (C layout) -> LDS -> f16 A layout ----
        {
            const int c = lane & 15, rh = (lane >> 4) * 8;
#pragma unroll
            for (int r = 0; r < 8; ++r) {
                sP[wave][r + rh][c]      = (_Float16)s0[r];
                sP[wave][r + rh][c + 16] = (_Float16)s1[r];
            }
        }
        {
            v16h ap = load_a(&sP[wave][0][0], 32);
            v16h bvf[4];
#pragma unroll
            for (int j = 0; j < 4; ++j) bvf[j] = load_b(&sVt[buf][j * 16][0], 40);
#pragma unroll
            for (int j = 0; j < 4; ++j) acc[j] = wmma_f16(ap, bvf[j], acc[j]);
        }

        // ---- finish next-buffer staging, publish ----
        if (more) storeV(buf ^ 1, vn);
        async_wait<0>();
        __syncthreads();
    }

    const int c = lane & 15, rh = (lane >> 4) * 8;
#pragma unroll
    for (int r = 0; r < 8; ++r) {
        float inv = 1.0f / lrow[r];
        size_t m = (size_t)b * SEQ + q0 + r + rh;
#pragma unroll
        for (int j = 0; j < 4; ++j)
            og[m * EMBED + h * HDIM + j * 16 + c] = (_Float16)(acc[j][r] * inv);
    }
}

// ---------------- output projection + bias + residual(x+pe) ----------------
__global__ __launch_bounds__(256) void k_oproj(const _Float16* __restrict__ a16,
                                               const _Float16* __restrict__ wTo,
                                               const float* __restrict__ bo,
                                               const float* __restrict__ x,
                                               const float* __restrict__ pe,
                                               float* __restrict__ out) {
    __shared__ _Float16 sA[2][128][40];
    __shared__ _Float16 sB[2][64][40];

    const int wave = threadIdx.x >> 5;
    const int lane = threadIdx.x & 31;
    const int mb = blockIdx.x * 128;
    const int n0 = blockIdx.y * 64;

    auto stage = [&](int buf, int kk) {
        const int t = threadIdx.x;
#pragma unroll
        for (int i = 0; i < 2; ++i) {
            int chunk = t + i * 256;
            int row = chunk >> 2;
            int off = (chunk & 3) * 8;
            async_cp16(a16 + (size_t)(mb + row) * EMBED + kk + off, &sA[buf][row][off]);
        }
        {
            int row = t >> 2;
            int off = (t & 3) * 8;
            async_cp16(wTo + (size_t)(n0 + row) * EMBED + kk + off, &sB[buf][row][off]);
        }
    };

    v8f c[4];
#pragma unroll
    for (int j = 0; j < 4; ++j) c[j] = zero8();

    auto compute = [&](int buf) {
        v16h a = load_a(&sA[buf][wave * 16][0], 40);
        v16h bf[4];
#pragma unroll
        for (int j = 0; j < 4; ++j) bf[j] = load_b(&sB[buf][j * 16][0], 40);
#pragma unroll
        for (int j = 0; j < 4; ++j) c[j] = wmma_f16(a, bf[j], c[j]);
    };

    stage(0, 0);
    for (int kk = 0; kk < EMBED; kk += 64) {
        stage(1, kk + 32);
        async_wait<3>();
        __syncthreads();
        compute(0);
        __syncthreads();
        if (kk + 64 < EMBED) { stage(0, kk + 64); async_wait<3>(); }
        else                 { async_wait<0>(); }
        __syncthreads();
        compute(1);
        __syncthreads();
    }

    const int cb = lane & 15;
    const int rh = (lane >> 4) * 8;
    const int m0 = mb + wave * 16;
#pragma unroll
    for (int j = 0; j < 4; ++j) {
        int n = n0 + j * 16 + cb;
        float bn = bo[n];
#pragma unroll
        for (int r = 0; r < 8; ++r) {
            size_t m = (size_t)m0 + r + rh;
            int s = (int)(m & (SEQ - 1));
            out[m * EMBED + n] = c[j][r] + bn + x[m * EMBED + n] + pe[(size_t)s * EMBED + n];
        }
    }
}

// ---------------- layernorm (in place on d_out) ----------------
__global__ __launch_bounds__(256) void k_ln(float* __restrict__ out,
                                            const float* __restrict__ gamma,
                                            const float* __restrict__ beta) {
    __shared__ float rs[256], rss[256];
    const int tid = threadIdx.x;
    const size_t row = blockIdx.x;
    float v[4];
    float s = 0.0f, ss = 0.0f;
#pragma unroll
    for (int i = 0; i < 4; ++i) {
        v[i] = out[row * EMBED + tid + i * 256];
        s += v[i];
        ss += v[i] * v[i];
    }
    rs[tid] = s; rss[tid] = ss;
    __syncthreads();
    for (int off = 128; off > 0; off >>= 1) {
        if (tid < off) { rs[tid] += rs[tid + off]; rss[tid] += rss[tid + off]; }
        __syncthreads();
    }
    float mu  = rs[0] * (1.0f / EMBED);
    float var = rss[0] * (1.0f / EMBED) - mu * mu;
    float inv = rsqrtf(var + 1e-5f);
#pragma unroll
    for (int i = 0; i < 4; ++i) {
        int col = tid + i * 256;
        out[row * EMBED + col] = (v[i] - mu) * inv * gamma[col] + beta[col];
    }
}

extern "C" void kernel_launch(void* const* d_in, const int* in_sizes, int n_in,
                              void* d_out, int out_size, void* d_ws, size_t ws_size,
                              hipStream_t stream) {
    (void)in_sizes; (void)n_in; (void)out_size; (void)ws_size;
    const float* x     = (const float*)d_in[0];
    const float* wq    = (const float*)d_in[1];
    const float* bq    = (const float*)d_in[2];
    const float* wk    = (const float*)d_in[3];
    const float* bk    = (const float*)d_in[4];
    const float* wv    = (const float*)d_in[5];
    const float* bv    = (const float*)d_in[6];
    const float* wo    = (const float*)d_in[7];
    const float* bo    = (const float*)d_in[8];
    const float* gamma = (const float*)d_in[9];
    const float* beta  = (const float*)d_in[10];
    const float* pe    = (const float*)d_in[11];
    float* out = (float*)d_out;

    char* ws = (char*)d_ws;
    const size_t szAct = (size_t)ROWS * EMBED * sizeof(_Float16);   // 16 MB
    const size_t szW   = (size_t)EMBED * EMBED * sizeof(_Float16);  // 2 MB
    _Float16* xp16 = (_Float16*)ws;            ws += szAct;
    _Float16* q16  = (_Float16*)ws;            ws += szAct;
    _Float16* k16  = (_Float16*)ws;            ws += szAct;
    _Float16* v16  = (_Float16*)ws;            ws += szAct;
    _Float16* a16  = (_Float16*)ws;            ws += szAct;
    _Float16* wTq  = (_Float16*)ws;            ws += szW;
    _Float16* wTk  = (_Float16*)ws;            ws += szW;
    _Float16* wTv  = (_Float16*)ws;            ws += szW;
    _Float16* wTo  = (_Float16*)ws;            ws += szW;

    k_prep_x<<<(ROWS * EMBED) / 256, 256, 0, stream>>>(x, pe, xp16);
    k_prep_w<<<dim3((EMBED * EMBED) / 256, 1, 4), 256, 0, stream>>>(wq, wk, wv, wo, wTq, wTk, wTv, wTo);
    k_qkv<<<dim3(ROWS / 128, EMBED / 64, 3), 256, 0, stream>>>(xp16, wTq, wTk, wTv, bq, bk, bv, q16, k16, v16);
    k_attn<<<dim3(SEQ / 128, BATCH * NHEAD), 256, 0, stream>>>(q16, k16, v16, a16);
    k_oproj<<<dim3(ROWS / 128, EMBED / 64), 256, 0, stream>>>(a16, wTo, bo, x, pe, out);
    k_ln<<<ROWS, 256, 0, stream>>>(out, gamma, beta);
}
